// MultiHeadAttention_69037304316012
// MI455X (gfx1250) — matrix operand order repaired and checked by reading, not compiled, not run on hardware
//
#include <hip/hip_runtime.h>

// ---------------------------------------------------------------------------
// Fused multi-head attention block for MI455X (gfx1250, wave32, WMMA).
// Pipeline: LN(q)->bf16 ; x@W GEMMs (v_wmma_f32_16x16x32_bf16, f32 accum) ;
// flash attention per (b,h,qtile) ; out@Wo + residual ; LN in place.
// ---------------------------------------------------------------------------

typedef __bf16 bf16_t;
typedef __attribute__((ext_vector_type(16))) __bf16 v16bf;
typedef __attribute__((ext_vector_type(8)))  __bf16 bf16x8;
typedef __attribute__((ext_vector_type(8)))  float   v8f;

__device__ __forceinline__ bf16_t f2bf(float f) {
    // round-to-nearest-even fp32 -> bf16
    unsigned u = __float_as_uint(f);
    unsigned r = (u + 0x7FFFu + ((u >> 16) & 1u)) >> 16;
    unsigned short s = (unsigned short)r;
    return __builtin_bit_cast(bf16_t, s);
}

__device__ __forceinline__ v16bf mkfrag(bf16x8 lo, bf16x8 hi) {
    return __builtin_shufflevector(lo, hi, 0,1,2,3,4,5,6,7,8,9,10,11,12,13,14,15);
}

// ---------------------------------------------------------------------------
// elementwise fp32 -> bf16
__global__ void cvt_f32_bf16(const float* __restrict__ in, bf16_t* __restrict__ out, int n) {
    int i = blockIdx.x * blockDim.x + threadIdx.x;
    int stride = gridDim.x * blockDim.x;
    for (; i < n; i += stride) out[i] = f2bf(in[i]);
}

// W[K][N] fp32 -> Wt[N][K] bf16 (so WMMA B-fragments load contiguous rows)
__global__ void transpose_to_bf16(const float* __restrict__ W, bf16_t* __restrict__ Wt,
                                  int K, int N) {
    int i = blockIdx.x * blockDim.x + threadIdx.x;
    if (i < K * N) {
        int k = i / N, n = i % N;
        Wt[(size_t)n * K + k] = f2bf(W[i]);
    }
}

// ---------------------------------------------------------------------------
// LayerNorm over rows of length D (=1024). 256 threads/row, float4 per thread.
// toBf16=1: write bf16 to outB.  toBf16=0: write fp32 to outF (in-place OK).
__global__ void __launch_bounds__(256)
layernorm_k(const float* __restrict__ x, float* __restrict__ outF,
            bf16_t* __restrict__ outB, const float* __restrict__ gamma,
            const float* __restrict__ beta, int D, int toBf16) {
    __shared__ float red_s[8], red_ss[8];
    const int row = blockIdx.x;
    const int tid = threadIdx.x;
    const float4* xr = (const float4*)(x + (size_t)row * D);
    float4 v = xr[tid];
    float s  = v.x + v.y + v.z + v.w;
    float ss = v.x * v.x + v.y * v.y + v.z * v.z + v.w * v.w;
#pragma unroll
    for (int m = 16; m >= 1; m >>= 1) {
        s  += __shfl_xor(s,  m, 32);
        ss += __shfl_xor(ss, m, 32);
    }
    const int wv = tid >> 5, lane = tid & 31;
    if (lane == 0) { red_s[wv] = s; red_ss[wv] = ss; }
    __syncthreads();
    float tot = 0.f, tot2 = 0.f;
#pragma unroll
    for (int j = 0; j < 8; j++) { tot += red_s[j]; tot2 += red_ss[j]; }
    const float mu   = tot / (float)D;
    const float var  = tot2 / (float)D - mu * mu;
    const float rstd = rsqrtf(var + 1e-6f);
    const int c0 = tid * 4;
    float y0 = (v.x - mu) * rstd * gamma[c0 + 0] + beta[c0 + 0];
    float y1 = (v.y - mu) * rstd * gamma[c0 + 1] + beta[c0 + 1];
    float y2 = (v.z - mu) * rstd * gamma[c0 + 2] + beta[c0 + 2];
    float y3 = (v.w - mu) * rstd * gamma[c0 + 3] + beta[c0 + 3];
    if (toBf16) {
        bf16_t* o = outB + (size_t)row * D + c0;
        o[0] = f2bf(y0); o[1] = f2bf(y1); o[2] = f2bf(y2); o[3] = f2bf(y3);
    } else {
        float4* o = (float4*)(outF + (size_t)row * D);
        o[tid] = make_float4(y0, y1, y2, y3);
    }
}

// ---------------------------------------------------------------------------
// C[M][N] = A[M][K] @ Bt[N][K]^T  via v_wmma_f32_16x16x32_bf16.
// Block = 256 thr (8 waves), block tile 256x64, wave tile 32x64
// (2 A-fragments per k-step, each B-fragment reused twice -> 8 WMMA/k-step).
// mode 0: Cb[M][N] bf16 row-major
// mode 1: head-transposed store Cb = Vt[b][h][d][l]  (assumes L=1024, H=16, Dh=64)
// mode 2: Cf[M][N] fp32 = acc + resid[M][N]
__global__ void __launch_bounds__(256)
gemm_bf16_wmma(const bf16_t* __restrict__ A, const bf16_t* __restrict__ Bt,
               bf16_t* __restrict__ Cb, float* __restrict__ Cf,
               const float* __restrict__ resid,
               int M, int N, int K, int mode) {
    __shared__ bf16_t ldsB[64][40];            // Bt tile: 64 n-rows x 32 k (pad 40)
    const int tid  = threadIdx.x;
    const int wv   = tid >> 5, lane = tid & 31;
    const int ln15 = lane & 15, hx = lane >> 4;
    const int a8   = hx * 8;                   // A-frag K start per lane half
    const int b16  = hx * 16;                  // unused; B-frag K start is a8 too
    const int blockN = blockIdx.x * 64;
    const int blockM = blockIdx.y * 256;
    const int rowA0  = blockM + wv * 32 + ln15;     // A fragment row, mi=0
    const int rowA1  = rowA0 + 16;                  // A fragment row, mi=1

    // staging assignment (one b128 per thread per k-step)
    const int stg_n = tid >> 2;                // 0..63
    const int stg_k = (tid & 3) * 8;           // 0,8,16,24
    const bf16_t* stg_src = Bt + (size_t)(blockN + stg_n) * K + stg_k;

    v8f acc[2][4] = {};

    for (int k0 = 0; k0 < K; k0 += 32) {
        // stage Bt tile: 256 threads x 8 bf16 each
        *(bf16x8*)&ldsB[stg_n][stg_k] = *(const bf16x8*)(stg_src + k0);
        // prefetch next tile's line into cache while WMMAs run
        if (k0 + 32 < K) __builtin_prefetch(stg_src + k0 + 32, 0, 3);
        __syncthreads();

        const bf16x8* ap0 = (const bf16x8*)(A + (size_t)rowA0 * K + k0 + a8);
        const bf16x8* ap1 = (const bf16x8*)(A + (size_t)rowA1 * K + k0 + a8);
        v16bf afrag0 = mkfrag(ap0[0], ap0[2]); // K = {a8..a8+7, a8+16..a8+23}
        v16bf afrag1 = mkfrag(ap1[0], ap1[2]);

#pragma unroll
        for (int nt = 0; nt < 4; nt++) {
            const bf16_t* bp = &ldsB[nt * 16 + ln15][a8];
            v16bf bfrag = mkfrag(*(const bf16x8*)bp, *(const bf16x8*)(bp + 16));
            acc[0][nt] = __builtin_amdgcn_wmma_f32_16x16x32_bf16(
                false, afrag0, false, bfrag, (short)0, acc[0][nt], false, false);
            acc[1][nt] = __builtin_amdgcn_wmma_f32_16x16x32_bf16(
                false, afrag1, false, bfrag, (short)0, acc[1][nt], false, false);
        }
        __syncthreads();
    }

#pragma unroll
    for (int mi = 0; mi < 2; mi++) {
#pragma unroll
        for (int nt = 0; nt < 4; nt++) {
#pragma unroll
            for (int r = 0; r < 8; r++) {
                int row = blockM + wv * 32 + mi * 16 + hx * 8 + r;
                int col = blockN + nt * 16 + ln15;
                float vv = acc[mi][nt][r];
                if (mode == 0) {
                    Cb[(size_t)row * N + col] = f2bf(vv);
                } else if (mode == 1) {      // Vt[((b*16+h)*64+d)][l], L=1024
                    int bb = row >> 10, l = row & 1023;
                    int h  = col >> 6,  d = col & 63;
                    Cb[(((size_t)(bb * 16 + h) * 64 + d) << 10) + l] = f2bf(vv);
                } else {
                    size_t idx = (size_t)row * N + col;
                    Cf[idx] = vv + resid[idx];
                }
            }
        }
    }
}

// ---------------------------------------------------------------------------
// Flash attention per (b, h, 64-row q block). 128 thr = 4 waves, 16 q rows/wave.
// Qp,Kp: [B*L][H*Dh] bf16 ; Vt: [B][H][Dh][L] bf16 ; ctx: [B*L][H*Dh] bf16.
__global__ void __launch_bounds__(128)
attn_wmma(const bf16_t* __restrict__ Qp, const bf16_t* __restrict__ Kp,
          const bf16_t* __restrict__ Vt, bf16_t* __restrict__ ctx,
          int Lq, int Lk) {
    __shared__ bf16_t ldsK[32][72];            // 32 keys x 64 d   (pad 72)
    __shared__ bf16_t ldsV[64][40];            // 64 d    x 32 key (pad 40)
    __shared__ bf16_t ldsP[4][16][40];         // per-wave P tile 16x32 (pad 40)

    const int tid  = threadIdx.x;
    const int wv   = tid >> 5, lane = tid & 31;
    const int ln15 = lane & 15, hx = lane >> 4;
    const int a8   = hx * 8;
    const int b16  = hx * 16;

    const int qblk = blockIdx.x & 15;          // Lq/64 = 16
    const int h    = (blockIdx.x >> 4) & 15;
    const int b    = blockIdx.x >> 8;
    const int HD   = 1024;                     // H*Dh
    const int q0   = qblk * 64 + wv * 16;

    // Q fragments for the two Dh k-steps (d0 = 0, 32); reused all key blocks
    const bf16_t* qrow = Qp + ((size_t)(b * Lq) + q0 + ln15) * HD + h * 64;
    v16bf aq[2];
#pragma unroll
    for (int j = 0; j < 2; j++) {
        const bf16x8* p = (const bf16x8*)(qrow + j * 32 + a8);
        aq[j] = mkfrag(p[0], p[2]);
    }

    v8f accO[4] = {};
    float mrow[8], lrow[8];
#pragma unroll
    for (int r = 0; r < 8; r++) { mrow[r] = -1e30f; lrow[r] = 0.f; }
    const float scl = 0.125f;                  // 1/sqrt(64)

    for (int key0 = 0; key0 < Lk; key0 += 32) {
        // stage K block (128 tasks) and V block (128 tasks)
        {
            int kl = tid >> 2;                 // 0..31
            int ds = (tid & 3) * 16;           // 0,16,32,48
            const bf16x8* sk = (const bf16x8*)(Kp + ((size_t)(b * Lk) + key0 + kl) * HD + h * 64 + ds);
            *(bf16x8*)&ldsK[kl][ds]     = sk[0];
            *(bf16x8*)&ldsK[kl][ds + 8] = sk[1];

            int dl = tid >> 1;                 // 0..63
            int ks = (tid & 1) * 16;           // 0,16
            const bf16x8* sv = (const bf16x8*)(Vt + ((size_t)((b * 16 + h) * 64 + dl)) * Lk + key0 + ks);
            *(bf16x8*)&ldsV[dl][ks]     = sv[0];
            *(bf16x8*)&ldsV[dl][ks + 8] = sv[1];
        }
        __syncthreads();

        // S tile 16 x 32 : two 16x16 N-tiles, Dh reduced in two K=32 steps
        v8f s[2] = {};
#pragma unroll
        for (int nt = 0; nt < 2; nt++) {
#pragma unroll
            for (int j = 0; j < 2; j++) {
                const bf16_t* kp = &ldsK[nt * 16 + ln15][j * 32 + a8];
                v16bf bk = mkfrag(*(const bf16x8*)kp, *(const bf16x8*)(kp + 16));
                s[nt] = __builtin_amdgcn_wmma_f32_16x16x32_bf16(
                    false, aq[j], false, bk, (short)0, s[nt], false, false);
            }
        }
#pragma unroll
        for (int nt = 0; nt < 2; nt++)
#pragma unroll
            for (int r = 0; r < 8; r++) s[nt][r] = s[nt][r] * scl;

        // online softmax per row (row = hx*8 + r lives in register r of this lane)
#pragma unroll
        for (int r = 0; r < 8; r++) {
            float rmax = fmaxf(s[0][r], s[1][r]);
#pragma unroll
            for (int m = 8; m >= 1; m >>= 1) rmax = fmaxf(rmax, __shfl_xor(rmax, m, 32));
            float mnew = fmaxf(mrow[r], rmax);
            float corr = __expf(mrow[r] - mnew);
            float p0 = __expf(s[0][r] - mnew);
            float p1 = __expf(s[1][r] - mnew);
            float psum = p0 + p1;
#pragma unroll
            for (int m = 8; m >= 1; m >>= 1) psum += __shfl_xor(psum, m, 32);
            lrow[r] = lrow[r] * corr + psum;
            mrow[r] = mnew;
#pragma unroll
            for (int nt = 0; nt < 4; nt++) accO[nt][r] = accO[nt][r] * corr;
            ldsP[wv][hx * 8 + r][ln15]      = f2bf(p0);   // C-layout -> LDS
            ldsP[wv][hx * 8 + r][16 + ln15] = f2bf(p1);
        }
        __syncthreads();

        // O += P(16x32) @ V(32x64)
        {
            const bf16_t* pp = &ldsP[wv][ln15][a8];
            v16bf ap2 = mkfrag(*(const bf16x8*)pp, *(const bf16x8*)(pp + 16));
#pragma unroll
            for (int nt = 0; nt < 4; nt++) {
                const bf16_t* vp = &ldsV[nt * 16 + ln15][a8];
                v16bf bv = mkfrag(*(const bf16x8*)vp, *(const bf16x8*)(vp + 16));
                accO[nt] = __builtin_amdgcn_wmma_f32_16x16x32_bf16(
                    false, ap2, false, bv, (short)0, accO[nt], false, false);
            }
        }
        __syncthreads();
    }

    // normalize and write context (concat heads) as bf16
#pragma unroll
    for (int r = 0; r < 8; r++) {
        float inv = 1.0f / lrow[r];
        int qr = q0 + hx * 8 + r;
#pragma unroll
        for (int nt = 0; nt < 4; nt++) {
            float vv = accO[nt][r] * inv;
            ctx[((size_t)(b * Lq) + qr) * HD + h * 64 + nt * 16 + ln15] = f2bf(vv);
        }
    }
}

// ---------------------------------------------------------------------------
extern "C" void kernel_launch(void* const* d_in, const int* in_sizes, int n_in,
                              void* d_out, int out_size, void* d_ws, size_t ws_size,
                              hipStream_t stream) {
    const float* q     = (const float*)d_in[0];
    const float* k     = (const float*)d_in[1];
    const float* v     = (const float*)d_in[2];
    const float* Wq    = (const float*)d_in[3];
    const float* Wk    = (const float*)d_in[4];
    const float* Wv    = (const float*)d_in[5];
    const float* Wo    = (const float*)d_in[6];
    const float* gamma = (const float*)d_in[7];
    const float* beta  = (const float*)d_in[8];
    float* out = (float*)d_out;

    const int B = 8, L = 1024, D = 1024, H = 16;
    const size_t ML  = (size_t)B * L;   // 8192 rows
    const size_t tok = ML * D;          // 8M elements

    bf16_t* ws  = (bf16_t*)d_ws;
    bf16_t* qn  = ws;  ws += tok;       // LN(q) bf16
    bf16_t* kx  = ws;  ws += tok;       // k bf16
    bf16_t* vx  = ws;  ws += tok;       // v bf16
    bf16_t* WqT = ws;  ws += (size_t)D * D;
    bf16_t* WkT = ws;  ws += (size_t)D * D;
    bf16_t* WvT = ws;  ws += (size_t)D * D;
    bf16_t* WoT = ws;  ws += (size_t)D * D;
    bf16_t* Qp  = ws;  ws += tok;       // [B*L][H*Dh]
    bf16_t* Kp  = ws;  ws += tok;       // [B*L][H*Dh]
    bf16_t* Vth = ws;  ws += tok;       // [B][H][Dh][L]
    bf16_t* ctx = ws;  ws += tok;       // [B*L][H*Dh]

    dim3 b256(256);

    // weights -> transposed bf16
    transpose_to_bf16<<<4096, b256, 0, stream>>>(Wq, WqT, D, D);
    transpose_to_bf16<<<4096, b256, 0, stream>>>(Wk, WkT, D, D);
    transpose_to_bf16<<<4096, b256, 0, stream>>>(Wv, WvT, D, D);
    transpose_to_bf16<<<4096, b256, 0, stream>>>(Wo, WoT, D, D);

    // k, v -> bf16 ; LN(q) -> bf16
    cvt_f32_bf16<<<2048, b256, 0, stream>>>(k, kx, (int)tok);
    cvt_f32_bf16<<<2048, b256, 0, stream>>>(v, vx, (int)tok);
    layernorm_k<<<(int)ML, b256, 0, stream>>>(q, nullptr, qn, gamma, beta, D, 1);

    // projections (WMMA GEMMs): grid = (N/64, M/256)
    dim3 ggrid(D / 64, (int)(ML / 256));
    gemm_bf16_wmma<<<ggrid, b256, 0, stream>>>(qn, WqT, Qp,  nullptr, nullptr, (int)ML, D, D, 0);
    gemm_bf16_wmma<<<ggrid, b256, 0, stream>>>(kx, WkT, Kp,  nullptr, nullptr, (int)ML, D, D, 0);
    gemm_bf16_wmma<<<ggrid, b256, 0, stream>>>(vx, WvT, Vth, nullptr, nullptr, (int)ML, D, D, 1);

    // flash attention: B*H*(L/64) blocks of 4 waves
    attn_wmma<<<B * H * (L / 64), dim3(128), 0, stream>>>(Qp, Kp, Vth, ctx, L, L);

    // output projection + residual (fp32), then LN in place -> d_out
    gemm_bf16_wmma<<<ggrid, b256, 0, stream>>>(ctx, WoT, nullptr, out, q, (int)ML, D, D, 2);
    layernorm_k<<<(int)ML, b256, 0, stream>>>(out, out, nullptr, gamma, beta, D, 0);
}
